// VariableBandSwinBlock_85770496901700
// MI455X (gfx1250) — matrix operand order repaired and checked by reading, not compiled, not run on hardware
//
#include <hip/hip_runtime.h>
#include <hip/hip_bf16.h>
#include <math.h>

#define DIM     96
#define HEADS   3
#define HD      32
#define HP      56
#define SHIFT   3
#define NT      196     // tokens per window (4 bands * 7*7)
#define NTP     208     // padded to 13*16
#define NTV     224     // padded to 7*32 (PV K dimension)
#define MT      13      // M tiles per window
#define LTOK    12544   // 4*56*56
#define MLP_HID 384
#define SLOT    7168    // u16 per (window,which,head) slot: max(208*32, 32*224)

typedef __attribute__((ext_vector_type(16))) __bf16 v16bf;
typedef __attribute__((ext_vector_type(8)))  float  v8f;

union VB16 { v16bf v; unsigned short u[16]; };

__device__ __forceinline__ unsigned short f2bf(float f) {
  union { float f; unsigned u; } t; t.f = f;
  unsigned u = t.u;
  u += 0x7FFFu + ((u >> 16) & 1u);   // round-to-nearest-even
  return (unsigned short)(u >> 16);
}
// A-matrix 16-bit K offset within a 32-wide K tile (ISA 7.12.2)
__device__ __forceinline__ int kofA(int j, int half) {
  return ((j >> 3) << 4) + (half << 3) + (j & 7);
}
// B-matrix 16-bit K offset within a 32-wide K tile
__device__ __forceinline__ int kofB(int j, int half) {
  return ((j >> 3) << 4) + (half << 3) + (j & 7);
}
// CDNA5 async DMA: global -> LDS, 16 bytes per lane, tracked by ASYNCcnt.
// Low 32 bits of a flat pointer to LDS are the LDS byte address (ISA 10.2).
__device__ __forceinline__ void async_cp16(unsigned lds_addr, const void* g) {
  asm volatile("global_load_async_to_lds_b128 %0, %1, off"
               :: "v"(lds_addr), "v"((unsigned long long)(uintptr_t)g)
               : "memory");
}
__device__ __forceinline__ void async_wait0() {
  asm volatile("s_wait_asynccnt 0x0" ::: "memory");
}

// ---------------- kernel 0: weights -> bf16, pre-transposed [N][K] ----------
__global__ void __launch_bounds__(256) k_prep(
    const float* __restrict__ qkv_w, const float* __restrict__ proj_w,
    const float* __restrict__ fc1_w, const float* __restrict__ fc2_w,
    unsigned short* __restrict__ wqT, unsigned short* __restrict__ wpT,
    unsigned short* __restrict__ w1T, unsigned short* __restrict__ w2T) {
  int i = blockIdx.x * blockDim.x + threadIdx.x;
  if (i < 288 * 96) {                       // qkv_w (96x288) -> [288][96]
    int n = i / 96, k = i - n * 96;
    wqT[i] = f2bf(qkv_w[k * 288 + n]);
  }
  if (i < 96 * 96) {                        // proj_w (96x96) -> [96][96]^T
    int n = i / 96, k = i - n * 96;
    wpT[i] = f2bf(proj_w[k * 96 + n]);
  }
  if (i < 384 * 96) {                       // fc1_w (96x384) -> [384][96]
    int n = i / 96, k = i - n * 96;
    w1T[i] = f2bf(fc1_w[k * 384 + n]);
  }
  if (i < 96 * 384) {                       // fc2_w (384x96) -> [96][384]
    int n = i / 384, k = i - n * 384;
    w2T[i] = f2bf(fc2_w[k * 96 + n]);
  }
}

// ---------------- kernel 1: shift + window gather + QKV GEMM ----------------
__global__ void __launch_bounds__(256) k_qkv(
    const float* __restrict__ x, const unsigned short* __restrict__ wqT,
    const float* __restrict__ qkv_b, unsigned short* __restrict__ qkvws) {
  __shared__ unsigned short Xbf[NTP * DIM];
  __shared__ int rowbase[NTP];
  const int w   = blockIdx.x;
  const int tid = threadIdx.x;
  const int wid = tid >> 5, lane = tid & 31;
  const int lr  = lane & 15, half = lane >> 4;

  if (tid < NTP) {
    int t = tid, rb = -1;
    if (t < NT) {
      int band = t / 49, p = t - band * 49;
      int wy = p / 7, wx = p - wy * 7;
      int bimg = w >> 6, wh = (w >> 3) & 7, ww = w & 7;
      int h  = wh * 7 + wy + SHIFT; if (h  >= HP) h  -= HP;
      int w2 = ww * 7 + wx + SHIFT; if (w2 >= HP) w2 -= HP;
      rb = (bimg * LTOK + band * (HP * HP) + h * HP + w2) * DIM;
    }
    rowbase[t] = rb;
  }
  if (tid < 96) __builtin_prefetch(wqT + tid * 288, 0, 3);
  __syncthreads();
  for (int i = tid; i < NTP * DIM; i += 256) {
    int m = i / DIM, c = i - m * DIM;
    int rb = rowbase[m];
    Xbf[i] = (rb >= 0) ? f2bf(x[rb + c]) : (unsigned short)0;
  }
  __syncthreads();

  for (int tt = wid; tt < MT * 18; tt += 8) {       // 208x96 @ 96x288
    int mt = tt % MT, nt = tt / MT;
    int m = mt * 16 + lr, n = nt * 16 + lr;
    v8f acc = {};
#pragma unroll
    for (int kt = 0; kt < 3; ++kt) {
      VB16 a, b;
#pragma unroll
      for (int j = 0; j < 16; ++j) {
        a.u[j] = Xbf[m * DIM + kt * 32 + kofA(j, half)];
        b.u[j] = wqT[(size_t)n * 96 + kt * 32 + kofB(j, half)];
      }
      acc = __builtin_amdgcn_wmma_f32_16x16x32_bf16(false, a.v, false, b.v,
                                                    (short)0, acc, false, false);
    }
    float bias = qkv_b[n];
    int which = n / 96, rem = n - which * 96;
    int head = rem >> 5, d = rem & 31;
    unsigned short* base = qkvws + (size_t)(w * 9 + which * 3 + head) * SLOT;
#pragma unroll
    for (int r = 0; r < 8; ++r) {
      int mm = mt * 16 + r + (half << 3);
      unsigned short vb = f2bf(acc[r] + bias);
      if (which < 2) base[mm * HD + d] = vb;        // Q,K row-major [tok][d]
      else           base[d * NTV + mm] = vb;       // V transposed  [d][tok]
    }
  }
}

// ---------------- kernel 2: window attention (per window,head) --------------
// Q/K/V slots are DMA'd into LDS with GLOBAL_LOAD_ASYNC_TO_LDS_B128 once per
// workgroup; all WMMA fragment gathers then hit LDS (ds_load_b128).
__global__ void __launch_bounds__(64) k_attn(
    const unsigned short* __restrict__ qkvws, unsigned short* __restrict__ ows) {
  __shared__ __align__(16) unsigned short Stage[20480]; // Q[6656] K[6656] V[7168]
  __shared__ float          Sf[2][16][NTP];
  __shared__ unsigned short Pb[2][16][NTV];
  const int w    = blockIdx.x / HEADS, head = blockIdx.x % HEADS;
  const int tid  = threadIdx.x;
  const int wid  = tid >> 5, lane = tid & 31;
  const int lr   = lane & 15, half = lane >> 4;
  const float scale = 0.17677669529663687f;         // 1/sqrt(32)

  {
    const char* gq = (const char*)(qkvws + (size_t)(w * 9 + 0 + head) * SLOT);
    const char* gk = (const char*)(qkvws + (size_t)(w * 9 + 3 + head) * SLOT);
    const char* gv = (const char*)(qkvws + (size_t)(w * 9 + 6 + head) * SLOT);
    unsigned lq = (unsigned)(uintptr_t)(&Stage[0]);
    unsigned lk = (unsigned)(uintptr_t)(&Stage[6656]);
    unsigned lv = (unsigned)(uintptr_t)(&Stage[13312]);
    for (int i = tid; i < 832; i += 64) {           // Q,K: 13312 B each
      async_cp16(lq + i * 16, gq + i * 16);
      async_cp16(lk + i * 16, gk + i * 16);
    }
    for (int i = tid; i < 896; i += 64)             // V: 14336 B
      async_cp16(lv + i * 16, gv + i * 16);
    async_wait0();
    __syncthreads();
  }
  const unsigned short* Qs = Stage;                 // [tok][32]
  const unsigned short* Ks = Stage + 6656;          // [tok][32]
  const unsigned short* Vs = Stage + 13312;         // [d][224]

  for (int mt = wid; mt < MT; mt += 2) {
    VB16 aq;
#pragma unroll
    for (int j = 0; j < 16; ++j)
      aq.u[j] = Qs[(mt * 16 + lr) * HD + kofA(j, half)];
    for (int nt = 0; nt < MT; ++nt) {               // S = Q K^T
      VB16 bk;
#pragma unroll
      for (int j = 0; j < 16; ++j)
        bk.u[j] = Ks[(nt * 16 + lr) * HD + kofB(j, half)];
      v8f acc = {};
      acc = __builtin_amdgcn_wmma_f32_16x16x32_bf16(false, aq.v, false, bk.v,
                                                    (short)0, acc, false, false);
#pragma unroll
      for (int r = 0; r < 8; ++r)
        Sf[wid][r + (half << 3)][nt * 16 + lr] = acc[r] * scale;
    }
    if (half == 0) {                                // fp32 softmax, row = lr
      int m = lr;
      float mx = -3.0e38f;
      for (int n = 0; n < NT; ++n) mx = fmaxf(mx, Sf[wid][m][n]);
      float s = 0.f;
      for (int n = 0; n < NT; ++n) s += __expf(Sf[wid][m][n] - mx);
      float inv = 1.0f / s;
      for (int n = 0; n < NT; ++n)
        Pb[wid][m][n] = f2bf(__expf(Sf[wid][m][n] - mx) * inv);
      for (int n = NT; n < NTV; ++n) Pb[wid][m][n] = 0;
    }
#pragma unroll
    for (int nt2 = 0; nt2 < 2; ++nt2) {             // O = P V  (16x32)
      v8f acc = {};
      for (int kt = 0; kt < 7; ++kt) {
        VB16 ap, bv;
#pragma unroll
        for (int j = 0; j < 16; ++j) {
          ap.u[j] = Pb[wid][lr][kt * 32 + kofA(j, half)];
          bv.u[j] = Vs[(nt2 * 16 + lr) * NTV + kt * 32 + kofB(j, half)];
        }
        acc = __builtin_amdgcn_wmma_f32_16x16x32_bf16(false, ap.v, false, bv.v,
                                                      (short)0, acc, false, false);
      }
#pragma unroll
      for (int r = 0; r < 8; ++r) {
        int mm = mt * 16 + r + (half << 3);
        ows[((size_t)w * NTP + mm) * DIM + head * HD + nt2 * 16 + lr] =
            f2bf(acc[r]);
      }
    }
  }
}

// ---------------- kernel 3: proj GEMM + window reverse scatter --------------
__global__ void __launch_bounds__(256) k_proj(
    const unsigned short* __restrict__ ows, const unsigned short* __restrict__ wpT,
    const float* __restrict__ proj_b, float* __restrict__ out) {
  const int w   = blockIdx.x;
  const int tid = threadIdx.x;
  const int wid = tid >> 5, lane = tid & 31;
  const int lr  = lane & 15, half = lane >> 4;
  const int bimg = w >> 6, wh = (w >> 3) & 7, ww = w & 7;
  for (int tt = wid; tt < MT * 6; tt += 8) {
    int mt = tt % MT, nt = tt / MT;
    v8f acc = {};
#pragma unroll
    for (int kt = 0; kt < 3; ++kt) {
      VB16 a, b;
#pragma unroll
      for (int j = 0; j < 16; ++j) {
        a.u[j] = ows[((size_t)w * NTP + mt * 16 + lr) * DIM + kt * 32 + kofA(j, half)];
        b.u[j] = wpT[(nt * 16 + lr) * 96 + kt * 32 + kofB(j, half)];
      }
      acc = __builtin_amdgcn_wmma_f32_16x16x32_bf16(false, a.v, false, b.v,
                                                    (short)0, acc, false, false);
    }
    int col = nt * 16 + lr;
    float bias = proj_b[col];
#pragma unroll
    for (int r = 0; r < 8; ++r) {
      int t = mt * 16 + r + (half << 3);
      if (t < NT) {                                 // roll indices self-inverse
        int band = t / 49, p = t - band * 49;
        int wy = p / 7, wx = p - wy * 7;
        int h  = wh * 7 + wy + SHIFT; if (h  >= HP) h  -= HP;
        int w2 = ww * 7 + wx + SHIFT; if (w2 >= HP) w2 -= HP;
        out[(size_t)(bimg * LTOK + band * (HP * HP) + h * HP + w2) * DIM + col] =
            acc[r] + bias;
      }
    }
  }
}

// ---------------- kernel 4: x2 = a+LN1(a); x2 += GELU(LN2(x2)W1+b)W2+b ------
__global__ void __launch_bounds__(256) k_mlp(
    float* __restrict__ out,
    const float* __restrict__ g1, const float* __restrict__ b1,
    const float* __restrict__ g2, const float* __restrict__ b2,
    const unsigned short* __restrict__ w1T, const float* __restrict__ fc1_b,
    const unsigned short* __restrict__ w2T, const float* __restrict__ fc2_b) {
  __shared__ float          Xf[32][DIM];
  __shared__ float          X2f[32][DIM];
  __shared__ unsigned short Abf[32][DIM];
  __shared__ unsigned short Hbf[32][MLP_HID];
  const int tid = threadIdx.x;
  const int wid = tid >> 5, lane = tid & 31;
  const int lr  = lane & 15, half = lane >> 4;
  const size_t t0 = (size_t)blockIdx.x * 32;

  if (tid < 128) __builtin_prefetch(w1T + (size_t)tid * 288, 0, 3);
  for (int i = tid; i < 32 * DIM; i += 256) {
    int m = i / DIM, c = i - m * DIM;
    Xf[m][c] = out[(t0 + m) * DIM + c];
  }
  __syncthreads();
  if (tid < 32) {
    int m = tid;
    float mu = 0.f;
    for (int c = 0; c < DIM; ++c) mu += Xf[m][c];
    mu *= (1.0f / DIM);
    float var = 0.f;
    for (int c = 0; c < DIM; ++c) { float d = Xf[m][c] - mu; var += d * d; }
    float rs = rsqrtf(var * (1.0f / DIM) + 1e-5f);
    float s2 = 0.f, q2 = 0.f;
    for (int c = 0; c < DIM; ++c) {
      float v  = Xf[m][c];
      float x2 = v + (v - mu) * rs * g1[c] + b1[c];
      X2f[m][c] = x2;
      s2 += x2; q2 += x2 * x2;
    }
    float mu2  = s2 * (1.0f / DIM);
    float var2 = q2 * (1.0f / DIM) - mu2 * mu2;
    float rs2  = rsqrtf(fmaxf(var2, 0.f) + 1e-5f);
    for (int c = 0; c < DIM; ++c)
      Abf[m][c] = f2bf((X2f[m][c] - mu2) * rs2 * g2[c] + b2[c]);
  }
  __syncthreads();
  for (int tt = wid; tt < 2 * 24; tt += 8) {        // H = gelu(A@W1 + b1)
    int mt = tt & 1, nt = tt >> 1;
    v8f acc = {};
#pragma unroll
    for (int kt = 0; kt < 3; ++kt) {
      VB16 a, b;
#pragma unroll
      for (int j = 0; j < 16; ++j) {
        a.u[j] = Abf[mt * 16 + lr][kt * 32 + kofA(j, half)];
        b.u[j] = w1T[(size_t)(nt * 16 + lr) * 96 + kt * 32 + kofB(j, half)];
      }
      acc = __builtin_amdgcn_wmma_f32_16x16x32_bf16(false, a.v, false, b.v,
                                                    (short)0, acc, false, false);
    }
    int col = nt * 16 + lr;
    float bias = fc1_b[col];
#pragma unroll
    for (int r = 0; r < 8; ++r) {
      float v = acc[r] + bias;
      v = 0.5f * v * (1.0f + erff(v * 0.70710678118654752f));
      Hbf[mt * 16 + r + (half << 3)][col] = f2bf(v);
    }
  }
  __syncthreads();
  for (int tt = wid; tt < 2 * 6; tt += 8) {         // out = X2 + H@W2 + b2
    int mt = tt & 1, nt = tt >> 1;
    v8f acc = {};
    for (int kt = 0; kt < 12; ++kt) {
      VB16 a, b;
#pragma unroll
      for (int j = 0; j < 16; ++j) {
        a.u[j] = Hbf[mt * 16 + lr][kt * 32 + kofA(j, half)];
        b.u[j] = w2T[(size_t)(nt * 16 + lr) * 384 + kt * 32 + kofB(j, half)];
      }
      acc = __builtin_amdgcn_wmma_f32_16x16x32_bf16(false, a.v, false, b.v,
                                                    (short)0, acc, false, false);
    }
    int col = nt * 16 + lr;
    float bias = fc2_b[col];
#pragma unroll
    for (int r = 0; r < 8; ++r) {
      int m = mt * 16 + r + (half << 3);
      out[(t0 + m) * DIM + col] = acc[r] + bias + X2f[m][col];
    }
  }
}

extern "C" void kernel_launch(void* const* d_in, const int* in_sizes, int n_in,
                              void* d_out, int out_size, void* d_ws, size_t ws_size,
                              hipStream_t stream) {
  (void)n_in; (void)out_size; (void)ws_size;
  const float* x      = (const float*)d_in[0];
  const float* qkv_w  = (const float*)d_in[1];
  const float* qkv_b  = (const float*)d_in[2];
  const float* proj_w = (const float*)d_in[3];
  const float* proj_b = (const float*)d_in[4];
  const float* g1     = (const float*)d_in[5];
  const float* b1     = (const float*)d_in[6];
  const float* g2     = (const float*)d_in[7];
  const float* b2     = (const float*)d_in[8];
  const float* fc1_w  = (const float*)d_in[9];
  const float* fc1_b  = (const float*)d_in[10];
  const float* fc2_w  = (const float*)d_in[11];
  const float* fc2_b  = (const float*)d_in[12];
  float* out = (float*)d_out;

  const int B    = in_sizes[0] / (LTOK * DIM);
  const int nwin = B * 64;

  unsigned short* wsu   = (unsigned short*)d_ws;
  unsigned short* wqT   = wsu;
  unsigned short* wpT   = wqT + 288 * 96;
  unsigned short* w1T   = wpT + 96 * 96;
  unsigned short* w2T   = w1T + 384 * 96;
  unsigned short* qkvws = w2T + 96 * 384;
  unsigned short* ows   = qkvws + (size_t)nwin * 9 * SLOT;

  k_prep<<<dim3(144), dim3(256), 0, stream>>>(qkv_w, proj_w, fc1_w, fc2_w,
                                              wqT, wpT, w1T, w2T);
  k_qkv<<<dim3(nwin), dim3(256), 0, stream>>>(x, wqT, qkv_b, qkvws);
  k_attn<<<dim3(nwin * HEADS), dim3(64), 0, stream>>>(qkvws, ows);
  k_proj<<<dim3(nwin), dim3(256), 0, stream>>>(ows, wpT, proj_b, out);
  k_mlp<<<dim3(B * (LTOK / 32)), dim3(256), 0, stream>>>(out, g1, b1, g2, b2,
                                                         w1T, fc1_b, w2T, fc2_b);
}
